// QRNN_2bit_44633300140404
// MI455X (gfx1250) — 8-bit matrix operand order repaired and checked by reading, compiled, not run on hardware
//
#include <hip/hip_runtime.h>
#include <stdint.h>


typedef int v8i __attribute__((ext_vector_type(8)));

static constexpr int B_ = 64, T_ = 512, E_ = 1024, H_ = 1024, L_ = 2, O_ = 4;
static constexpr long V_ = 30000;
static constexpr int BT_ = B_ * T_;
static constexpr float WTHR = 0.05f;
static constexpr float ATHR = 0.5f;
static constexpr float WS = 0.1f;

static constexpr size_t ALGN = 256;
constexpr size_t align_up(size_t x) { return (x + ALGN - 1) & ~(ALGN - 1); }

static constexpr size_t OFF_ABSMAX = 0;
static constexpr size_t OFF_SCALE  = 64;
static constexpr size_t OFF_CNT    = 128;
static constexpr size_t OFF_HBUF   = align_up(OFF_CNT + (size_t)L_ * T_ * 4);
static constexpr size_t HBUF_SZ    = (size_t)L_ * 2 * B_ * H_;
static constexpr size_t ZERO_BYTES = OFF_HBUF + HBUF_SZ;
static constexpr size_t OFF_WIQ    = align_up(ZERO_BYTES);
static constexpr size_t OFF_WHQ    = OFF_WIQ + (size_t)H_ * E_;
static constexpr size_t OFF_FCQ    = OFF_WHQ + (size_t)H_ * H_;
static constexpr size_t OFF_BIAS   = align_up(OFF_FCQ + (size_t)O_ * H_);
static constexpr size_t OFF_XA     = align_up(OFF_BIAS + (size_t)H_ * 4);
static constexpr size_t OFF_XB     = OFF_XA + (size_t)BT_ * E_;
static constexpr size_t OFF_XW     = OFF_XB + (size_t)BT_ * E_;

__device__ __forceinline__ v8i load_a64(const signed char* p) {
  int2 c0 = *(const int2*)(p +  0);
  int2 c1 = *(const int2*)(p + 16);
  int2 c2 = *(const int2*)(p + 32);
  int2 c3 = *(const int2*)(p + 48);
  v8i a; a[0]=c0.x; a[1]=c0.y; a[2]=c1.x; a[3]=c1.y;
         a[4]=c2.x; a[5]=c2.y; a[6]=c3.x; a[7]=c3.y;
  return a;
}
__device__ __forceinline__ v8i load_b64(const signed char* p) {
  int2 c0 = *(const int2*)(p +  0);
  int2 c1 = *(const int2*)(p + 16);
  int2 c2 = *(const int2*)(p + 32);
  int2 c3 = *(const int2*)(p + 48);
  v8i b; b[0]=c0.x; b[1]=c0.y; b[2]=c1.x; b[3]=c1.y;
         b[4]=c2.x; b[5]=c2.y; b[6]=c3.x; b[7]=c3.y;
  return b;
}

__global__ void k_init(int* ws, long nwords) {
  long stride = (long)gridDim.x * blockDim.x;
  for (long i = (long)blockIdx.x * blockDim.x + threadIdx.x; i < nwords; i += stride)
    ws[i] = 0;
}

__global__ void k_absmax(const float* __restrict__ w, long n, unsigned* out) {
  float m = 0.f;
  long stride = (long)gridDim.x * blockDim.x;
  for (long i = (long)blockIdx.x * blockDim.x + threadIdx.x; i < n; i += stride)
    m = fmaxf(m, fabsf(w[i]));
  __shared__ float red[256];
  red[threadIdx.x] = m;
  __syncthreads();
  for (int s = blockDim.x >> 1; s > 0; s >>= 1) {
    if ((int)threadIdx.x < s) red[threadIdx.x] = fmaxf(red[threadIdx.x], red[threadIdx.x + s]);
    __syncthreads();
  }
  if (threadIdx.x == 0) atomicMax(out, __float_as_uint(red[0]));
}

__global__ void k_scale(const unsigned* am, float* sc) {
  float m = __uint_as_float(*am);
  *sc = (m > 0.f) ? exp2f(ceilf(log2f(m))) : 1.f;
}

__global__ void k_tern(const float* __restrict__ w, signed char* __restrict__ q,
                       int n, float thr) {
  int i = blockIdx.x * blockDim.x + threadIdx.x;
  if (i < n) {
    float v = w[i];
    q[i] = v > thr ? (signed char)1 : (v < -thr ? (signed char)-1 : (signed char)0);
  }
}

__global__ void k_bias(const float* bi, const float* bh, float* b, int n) {
  int i = blockIdx.x * blockDim.x + threadIdx.x;
  if (i < n) b[i] = bi[i] + bh[i];
}

__global__ void k_embed(const int* __restrict__ text, const float* __restrict__ emb,
                        const float* __restrict__ scp, signed char* __restrict__ x) {
  int i = blockIdx.x * blockDim.x + threadIdx.x;
  if (i >= BT_ * (E_ / 4)) return;
  int e4 = i % (E_ / 4);
  int bt = i / (E_ / 4);
  int b  = bt % B_;
  int t  = bt / B_;
  int tok = text[b * T_ + t];
  float s = *scp;
  const float* src = emb + (size_t)tok * E_ + e4 * 4;
  unsigned pack = 0;
  #pragma unroll
  for (int j = 0; j < 4; ++j) {
    float r = rintf(src[j] / s);
    r = fminf(fmaxf(r, -2.f), 1.f);
    float qv = r * s;
    signed char tq = qv > ATHR ? 1 : (qv < -ATHR ? -1 : 0);
    pack |= ((unsigned)(unsigned char)tq) << (8 * j);
  }
  *(unsigned*)(x + (size_t)bt * E_ + e4 * 4) = pack;
}

__global__ void k_gemm_xw(const signed char* __restrict__ x,
                          const signed char* __restrict__ wiq,
                          const float* __restrict__ bias,
                          float* __restrict__ xw) {
  int lane = threadIdx.x & 31;
  int wave = threadIdx.x >> 5;
  int r0 = blockIdx.x * 16;
  int n0 = blockIdx.y * 64 + wave * 16;
  const signed char* ap = x   + (size_t)(r0 + (lane & 15)) * E_ + ((lane >> 4) ? 8 : 0);
  const signed char* bp = wiq + (size_t)(n0 + (lane & 15)) * E_ + ((lane >> 4) ? 8 : 0);
  v8i acc0 = {}, acc1 = {};
  #pragma unroll
  for (int k = 0; k < E_; k += 128) {
    v8i a0 = load_a64(ap + k);
    v8i b0 = load_b64(bp + k);
    acc0 = __builtin_amdgcn_wmma_i32_16x16x64_iu8(true, a0, true, b0, acc0, false, false);
    v8i a1 = load_a64(ap + k + 64);
    v8i b1 = load_b64(bp + k + 64);
    acc1 = __builtin_amdgcn_wmma_i32_16x16x64_iu8(true, a1, true, b1, acc1, false, false);
  }
  int col   = n0 + (lane & 15);
  int msel  = (lane >> 4) * 8;
  int t     = r0 / B_;
  int bbase = (r0 % B_) + msel;
  float bb = bias[col];
  float* dst = xw + ((size_t)t * H_ + col) * B_ + bbase;
  float4 o0, o1;
  o0.x = WS * (float)(acc0[0] + acc1[0]) + bb; o0.y = WS * (float)(acc0[1] + acc1[1]) + bb;
  o0.z = WS * (float)(acc0[2] + acc1[2]) + bb; o0.w = WS * (float)(acc0[3] + acc1[3]) + bb;
  o1.x = WS * (float)(acc0[4] + acc1[4]) + bb; o1.y = WS * (float)(acc0[5] + acc1[5]) + bb;
  o1.z = WS * (float)(acc0[6] + acc1[6]) + bb; o1.w = WS * (float)(acc0[7] + acc1[7]) + bb;
  *(float4*)(dst)     = o0;
  *(float4*)(dst + 4) = o1;
}

__global__ void k_recurrent(const float* __restrict__ xw,
                            const signed char* __restrict__ whq,
                            signed char* __restrict__ hping,
                            signed char* __restrict__ xnext,
                            int* __restrict__ cnt, int nblk) {
  __shared__ signed char lwh[16 * H_];
  const int n0 = blockIdx.x * 16;

  for (int i = threadIdx.x * 16; i < 16 * H_; i += blockDim.x * 16) {
    unsigned dst = (unsigned)(uintptr_t)(lwh + i);
    const signed char* src = whq + (size_t)n0 * H_ + i;
    asm volatile("global_load_async_to_lds_b128 %0, %1, off"
                 :: "v"(dst), "v"(src) : "memory");
  }
  asm volatile("s_wait_asynccnt 0x0" ::: "memory");
  __syncthreads();

  const int lane = threadIdx.x & 31;
  const int wave = threadIdx.x >> 5;
  const int r0   = wave * 16;
  const signed char* bp = lwh + (lane & 15) * H_ + ((lane >> 4) ? 8 : 0);
  const int col  = n0 + (lane & 15);
  const int msel = (lane >> 4) * 8;
  const int rb   = r0 + msel;

  for (int t = 0; t < T_; ++t) {
    const signed char* hin  = hping + (size_t)(t & 1) * B_ * H_;
    signed char*       hout = hping + (size_t)((t + 1) & 1) * B_ * H_;
    const signed char* ap = hin + (size_t)(r0 + (lane & 15)) * H_ + ((lane >> 4) ? 8 : 0);

    v8i acc0 = {}, acc1 = {};
    #pragma unroll
    for (int k = 0; k < H_; k += 128) {
      v8i a0 = load_a64(ap + k);
      v8i b0 = load_b64(bp + k);
      acc0 = __builtin_amdgcn_wmma_i32_16x16x64_iu8(true, a0, true, b0, acc0, false, false);
      v8i a1 = load_a64(ap + k + 64);
      v8i b1 = load_b64(bp + k + 64);
      acc1 = __builtin_amdgcn_wmma_i32_16x16x64_iu8(true, a1, true, b1, acc1, false, false);
    }

    const float* xr = xw + ((size_t)t * H_ + col) * B_ + rb;
    if (t + 1 < T_)
      __builtin_prefetch(xw + ((size_t)(t + 1) * H_ + col) * B_ + rb, 0, 1);
    float4 x0 = *(const float4*)(xr);
    float4 x1 = *(const float4*)(xr + 4);
    float px[8] = {x0.x, x0.y, x0.z, x0.w, x1.x, x1.y, x1.z, x1.w};

    signed char hv[8];
    #pragma unroll
    for (int j = 0; j < 8; ++j) {
      float pre = px[j] + WS * (float)(acc0[j] + acc1[j]);
      hv[j] = pre > ATHR ? 1 : (pre < -ATHR ? -1 : 0);
    }
    #pragma unroll
    for (int j = 0; j < 8; ++j) {
      hout[(size_t)(rb + j) * H_ + col]           = hv[j];
      xnext[((size_t)t * B_ + rb + j) * H_ + col] = hv[j];
    }

    __threadfence();
    __syncthreads();
    if (threadIdx.x == 0) {
      __hip_atomic_fetch_add(&cnt[t], 1, __ATOMIC_RELEASE, __HIP_MEMORY_SCOPE_AGENT);
      while (__hip_atomic_load(&cnt[t], __ATOMIC_ACQUIRE, __HIP_MEMORY_SCOPE_AGENT) < nblk)
        __builtin_amdgcn_s_sleep(1);
    }
    __syncthreads();
    __builtin_amdgcn_fence(__ATOMIC_ACQUIRE, "agent");
  }
}

__global__ void k_fc(const signed char* __restrict__ hl0,
                     const signed char* __restrict__ hl1,
                     const signed char* __restrict__ fcq,
                     float* __restrict__ out) {
  int i = blockIdx.x * blockDim.x + threadIdx.x;
  if (i >= L_ * B_ * O_) return;
  int o = i % O_;
  int b = (i / O_) % B_;
  int l = i / (O_ * B_);
  const signed char* h = (l == 0 ? hl0 : hl1) + ((size_t)(T_ - 1) * B_ + b) * H_;
  const signed char* f = fcq + (size_t)o * H_;
  int s = 0;
  for (int k = 0; k < H_; ++k) s += (int)h[k] * (int)f[k];
  out[i] = WS * (float)s;
}

extern "C" void kernel_launch(void* const* d_in, const int* in_sizes, int n_in,
                              void* d_out, int out_size, void* d_ws, size_t ws_size,
                              hipStream_t stream) {
  (void)in_sizes; (void)n_in; (void)out_size; (void)ws_size;
  const int*   text = (const int*)  d_in[0];
  const float* emb  = (const float*)d_in[2];
  const float* W_ih = (const float*)d_in[3];
  const float* W_hh = (const float*)d_in[4];
  const float* b_ih = (const float*)d_in[5];
  const float* b_hh = (const float*)d_in[6];
  const float* fc_W = (const float*)d_in[7];
  float* out = (float*)d_out;

  char* ws = (char*)d_ws;
  unsigned*    absmax = (unsigned*)(ws + OFF_ABSMAX);
  float*       scale  = (float*)   (ws + OFF_SCALE);
  int*         cnt    = (int*)     (ws + OFF_CNT);
  signed char* hbuf   = (signed char*)(ws + OFF_HBUF);
  signed char* wiq    = (signed char*)(ws + OFF_WIQ);
  signed char* whq    = (signed char*)(ws + OFF_WHQ);
  signed char* fcq    = (signed char*)(ws + OFF_FCQ);
  float*       bias   = (float*)   (ws + OFF_BIAS);
  signed char* xa     = (signed char*)(ws + OFF_XA);
  signed char* xb     = (signed char*)(ws + OFF_XB);
  float*       xw     = (float*)   (ws + OFF_XW);

  k_init<<<256, 256, 0, stream>>>((int*)ws, (long)(ZERO_BYTES / 4));
  k_absmax<<<2048, 256, 0, stream>>>(emb, V_ * (long)E_, absmax);
  k_scale<<<1, 1, 0, stream>>>(absmax, scale);
  k_tern<<<(O_ * H_ + 255) / 256, 256, 0, stream>>>(fc_W, fcq, O_ * H_, WTHR);
  k_embed<<<(BT_ * (E_ / 4) + 255) / 256, 256, 0, stream>>>(text, emb, scale, xa);

  for (int l = 0; l < L_; ++l) {
    const signed char* xin  = l ? xb : xa;
    signed char*       xout = l ? xa : xb;
    k_tern<<<(H_ * E_ + 255) / 256, 256, 0, stream>>>(W_ih + (size_t)l * H_ * E_, wiq, H_ * E_, WTHR);
    k_tern<<<(H_ * H_ + 255) / 256, 256, 0, stream>>>(W_hh + (size_t)l * H_ * H_, whq, H_ * H_, WTHR);
    k_bias<<<(H_ + 255) / 256, 256, 0, stream>>>(b_ih + (size_t)l * H_, b_hh + (size_t)l * H_, bias, H_);
    k_gemm_xw<<<dim3(BT_ / 16, H_ / 64), 128, 0, stream>>>(xin, wiq, bias, xw);
    k_recurrent<<<H_ / 16, 128, 0, stream>>>(xw, whq,
                                             hbuf + (size_t)l * 2 * B_ * H_,
                                             xout, cnt + (size_t)l * T_, H_ / 16);
  }
  k_fc<<<2, 256, 0, stream>>>(xb, xa, fcq, out);
}
